// GATCritic_20126216749449
// MI455X (gfx1250) — matrix operand order repaired and checked by reading, not compiled, not run on hardware
//
#include <hip/hip_runtime.h>
#include <hip/hip_bf16.h>
#include <stdint.h>

#define NN 50000
#define EE 400000
#define HC 256
#define Gg 64
#define NEG_SLOPE 0.2f

typedef __attribute__((ext_vector_type(16))) __bf16 v16bf;
typedef __attribute__((ext_vector_type(8)))  float  v8f;

union Frag { v16bf v; uint4 q[2]; };

__device__ __forceinline__ unsigned short f2bf(float f) {
  union { float f; uint32_t u; } c; c.f = f;
  uint32_t r = c.u + 0x7fffu + ((c.u >> 16) & 1u);   // round-to-nearest-even
  return (unsigned short)(r >> 16);
}
__device__ __forceinline__ int fenc(float f) {        // order-preserving float->int
  int i = __float_as_int(f);
  return (i >= 0) ? i : (i ^ 0x7fffffff);
}
__device__ __forceinline__ float fdec(int i) {
  return __int_as_float((i >= 0) ? i : (i ^ 0x7fffffff));
}

// ---------------- generic fill ----------------
__global__ void k_fill(uint32_t* p, uint32_t v, int n) {
  int i = blockIdx.x * 256 + threadIdx.x;
  if (i < n) p[i] = v;
}

// ---------------- W (256x256 f32, [k][n]) -> Wt bf16 [n][k] ----------------
__global__ void k_wt(const float* W, unsigned short* Wt) {
  int nv = blockIdx.x;          // output row (n)
  int k  = threadIdx.x;         // 256 threads
  Wt[nv * 256 + k] = f2bf(W[k * 256 + nv]);
}

// ---------------- layer-1 GEMM: [N x 4] @ [4 x 256] ----------------
__global__ void k_gemm_in4(const float* x, const float* W, float* Hout) {
  int i = blockIdx.x, c = threadIdx.x;
  float a = 0.f;
#pragma unroll
  for (int k = 0; k < 4; ++k) a += x[i * 4 + k] * W[k * 256 + c];
  Hout[(size_t)i * 256 + c] = a;
}

// ---------------- bf16 WMMA GEMM: [nrows x 256] @ [256 x 256] ----------------
// A row-major bf16, Wt row-major bf16 [n][k] (W transposed). 4 waves/block,
// each wave computes a 16(M) x 64(N) tile. Per k-step: issue A + all four
// B fragment loads up front (separate registers) so the compiler can keep a
// staggered s_wait_loadcnt pipeline instead of load->wait0->wmma serialization.
__global__ void __launch_bounds__(128) k_gemm_wmma(const unsigned short* A,
                                                   const unsigned short* Wt,
                                                   float* Hout, int nrows) {
  int w    = threadIdx.x >> 5;
  int lane = threadIdx.x & 31;
  int m0   = blockIdx.x * 64 + w * 16;
  int n0   = blockIdx.y * 64;

  int row  = m0 + (lane & 15);
  row = (row < nrows) ? row : (nrows - 1);            // clamp loads; stores predicated
  int kbA  = (lane >> 4) * 16;                        // A: contiguous K-16 halves
  int kbB  = (lane >> 4) * 16;                        // B: contiguous K-16 halves

  const unsigned short* ar = A  + (size_t)row * 256 + kbA;
  const unsigned short* br = Wt + (size_t)(n0 + (lane & 15)) * 256 + kbB;

  v8f acc0 = {}, acc1 = {}, acc2 = {}, acc3 = {};
#pragma unroll
  for (int ks = 0; ks < 256; ks += 32) {
    Frag a, b0, b1, b2, b3;
    a.q[0]  = *(const uint4*)(ar + ks);
    a.q[1]  = *(const uint4*)(ar + ks + 8);
    b0.q[0] = *(const uint4*)(br + ks);
    b0.q[1] = *(const uint4*)(br + ks + 8);
    b1.q[0] = *(const uint4*)(br + 16 * 256 + ks);
    b1.q[1] = *(const uint4*)(br + 16 * 256 + ks + 8);
    b2.q[0] = *(const uint4*)(br + 32 * 256 + ks);
    b2.q[1] = *(const uint4*)(br + 32 * 256 + ks + 8);
    b3.q[0] = *(const uint4*)(br + 48 * 256 + ks);
    b3.q[1] = *(const uint4*)(br + 48 * 256 + ks + 8);
    acc0 = __builtin_amdgcn_wmma_f32_16x16x32_bf16(false, a.v, false, b0.v, (short)0, acc0, false, false);
    acc1 = __builtin_amdgcn_wmma_f32_16x16x32_bf16(false, a.v, false, b1.v, (short)0, acc1, false, false);
    acc2 = __builtin_amdgcn_wmma_f32_16x16x32_bf16(false, a.v, false, b2.v, (short)0, acc2, false, false);
    acc3 = __builtin_amdgcn_wmma_f32_16x16x32_bf16(false, a.v, false, b3.v, (short)0, acc3, false, false);
  }

  // D layout: lanes 0-15 -> M=r, lanes 16-31 -> M=8+r; N = lane&15
  int colb = n0 + (lane & 15);
  int rb   = m0 + ((lane & 16) ? 8 : 0);
#pragma unroll
  for (int r = 0; r < 8; ++r) {
    int rr = rb + r;
    if (rr < nrows) {
      Hout[(size_t)rr * 256 + colb     ] = acc0[r];
      Hout[(size_t)rr * 256 + colb + 16] = acc1[r];
      Hout[(size_t)rr * 256 + colb + 32] = acc2[r];
      Hout[(size_t)rr * 256 + colb + 48] = acc3[r];
    }
  }
}

// ---------------- per-node attention scores a_s, a_d ----------------
__global__ void k_att(const float* Hm, const float* asr, const float* ads,
                      float* as_, float* ad_, int n) {
  int id = blockIdx.x * 256 + threadIdx.x;
  if (id >= n * 8) return;
  int i = id >> 3, hh = id & 7;
  const float* hp = Hm + (size_t)i * 256 + hh * 32;
  const float* s1 = asr + hh * 32;
  const float* s2 = ads + hh * 32;
  float a = 0.f, d = 0.f;
#pragma unroll
  for (int c = 0; c < 32; ++c) { float v = hp[c]; a += v * s1[c]; d += v * s2[c]; }
  as_[id] = a; ad_[id] = d;
}

// ---------------- per-head edge scalar: s[h] = sum_c W_e[h*32+c]*att_edge[h,c] ----
__global__ void k_svec(const float* We, const float* ae, float* s) {
  int hh = threadIdx.x;
  if (hh < 8) {
    float a = 0.f;
    for (int c = 0; c < 32; ++c) a += We[hh * 32 + c] * ae[hh * 32 + c];
    s[hh] = a;
  }
}

// ---------------- self-loop prep: deg + edge_attr sum per dst ----------------
__global__ void k_degsum(const int* dst, const float* ea, float* deg, float* sea, int E) {
  int j = blockIdx.x * 256 + threadIdx.x;
  if (j < E) {
    int d = dst[j];
    atomicAdd(&deg[d], 1.0f);
    atomicAdd(&sea[d], ea[j]);
  }
}
__global__ void k_loopattr(const float* deg, const float* sea, float* la, int n) {
  int i = blockIdx.x * 256 + threadIdx.x;
  if (i < n) la[i] = sea[i] / fmaxf(deg[i], 1.0f);
}

// ---------------- logits + segment-max (float atomicMax via int encoding) -----
__global__ void k_logits(const int* src, const int* dst, const float* ea, const float* la,
                         const float* as_, const float* ad_, const float* sv,
                         float* zb, int* menc, int E, int nloop) {
  int j = blockIdx.x * 256 + threadIdx.x;
  if (j >= E + nloop) return;
  int s_, d_; float av;
  if (j < E) { s_ = src[j]; d_ = dst[j]; av = ea[j]; }
  else       { s_ = d_ = j - E; av = la[j - E]; }
#pragma unroll
  for (int hh = 0; hh < 8; ++hh) {
    float l = as_[s_ * 8 + hh] + ad_[d_ * 8 + hh] + av * sv[hh];
    l = (l > 0.f) ? l : NEG_SLOPE * l;
    zb[(size_t)j * 8 + hh] = l;
    atomicMax(&menc[d_ * 8 + hh], fenc(l));
  }
}

// ---------------- exp(logit - max) + segment denom ----------------
__global__ void k_expz(const int* dst, float* zb, const int* menc, float* den,
                       int E, int nloop) {
  int j = blockIdx.x * 256 + threadIdx.x;
  if (j >= E + nloop) return;
  int d_ = (j < E) ? dst[j] : (j - E);
#pragma unroll
  for (int hh = 0; hh < 8; ++hh) {
    float z = __expf(zb[(size_t)j * 8 + hh] - fdec(menc[d_ * 8 + hh]));
    zb[(size_t)j * 8 + hh] = z;
    atomicAdd(&den[d_ * 8 + hh], z);
  }
}

// ---------------- weighted scatter: out[dst] += h[src]*alpha ----------------
__global__ void k_scatter(const int* src, const int* dst, const float* zb, const float* den,
                          const float* Hm, float* xout, int E) {
  int j = blockIdx.x;
  int t = threadIdx.x;
  int s_, d_;
  if (j < E) { s_ = src[j]; d_ = dst[j]; }
  else       { s_ = d_ = j - E; }
  int hh = t >> 5;
  float alpha = zb[(size_t)j * 8 + hh] / (den[d_ * 8 + hh] + 1e-16f);
  atomicAdd(&xout[(size_t)d_ * 256 + t], Hm[(size_t)s_ * 256 + t] * alpha);
}

// ---------------- bias + relu, plus bf16 copy for next layer's GEMM ----------
__global__ void k_biasrelu(float* x, const float* b, unsigned short* abf, int n) {
  int id = blockIdx.x * 256 + threadIdx.x;
  if (id >= n) return;
  float v = fmaxf(x[id] + b[id & 255], 0.f);
  x[id] = v;
  abf[id] = f2bf(v);
}

// ---------------- global mean pool (sum + count) ----------------
__global__ void k_pool(const float* x, const int* bat, float* gsum, float* gcnt) {
  int i = blockIdx.x, t = threadIdx.x;
  int b = bat[i];
  atomicAdd(&gsum[b * 256 + t], x[(size_t)i * 256 + t]);
  if (t == 0) atomicAdd(&gcnt[b], 1.0f);
}

// ---------------- head: mean, fc1+relu, fc2 ----------------
__global__ void __launch_bounds__(128) k_head(const float* gsum, const float* gcnt,
                                              const float* W1, const float* b1,
                                              const float* W2, const float* b2,
                                              float* out) {
  __shared__ float g[256];
  __shared__ float red[128];
  int b = blockIdx.x, t = threadIdx.x;
  float cnt = fmaxf(gcnt[b], 1.f);
  for (int c = t; c < 256; c += 128) g[c] = gsum[b * 256 + c] / cnt;
  __syncthreads();
  float acc = b1[t];
  for (int c = 0; c < 256; ++c) acc += g[c] * W1[c * 128 + t];
  red[t] = fmaxf(acc, 0.f) * W2[t];
  __syncthreads();
  for (int s = 64; s > 0; s >>= 1) {
    if (t < s) red[t] += red[t + s];
    __syncthreads();
  }
  if (t == 0) out[b] = red[0] + b2[0];
}

extern "C" void kernel_launch(void* const* d_in, const int* in_sizes, int n_in,
                              void* d_out, int out_size, void* d_ws, size_t ws_size,
                              hipStream_t stream) {
  (void)in_sizes; (void)n_in; (void)out_size; (void)ws_size;

  const float* d_x  = (const float*)d_in[0];
  const int*   src  = (const int*)d_in[1];        // edge_index[0]
  const int*   dst  = src + EE;                   // edge_index[1]
  const float* ea   = (const float*)d_in[2];
  const int*   bat  = (const int*)d_in[3];
  // params flattened in dict order: l1..l3 {W, att_src, att_dst, W_e, att_edge, b}
  auto Lp = [&](int l, int k) { return (const float*)d_in[4 + 6 * l + k]; };
  const float* fc1W = (const float*)d_in[22];
  const float* fc1b = (const float*)d_in[23];
  const float* fc2W = (const float*)d_in[24];
  const float* fc2b = (const float*)d_in[25];

  // workspace carve (256B aligned)
  char* wsp = (char*)d_ws; size_t off = 0;
  auto carve = [&](size_t bytes) -> void* {
    void* p = wsp + off; off += (bytes + 255) & ~(size_t)255; return p;
  };
  float*          xbuf = (float*)carve((size_t)NN * 256 * 4);   // x / scatter output
  float*          hbuf = (float*)carve((size_t)NN * 256 * 4);   // transformed features
  unsigned short* abf  = (unsigned short*)carve((size_t)NN * 256 * 2); // bf16 activations
  unsigned short* wt   = (unsigned short*)carve((size_t)256 * 256 * 2); // bf16 W^T
  float*          as_  = (float*)carve((size_t)NN * 8 * 4);
  float*          ad_  = (float*)carve((size_t)NN * 8 * 4);
  int*            menc = (int*)carve((size_t)NN * 8 * 4);
  float*          den  = (float*)carve((size_t)NN * 8 * 4);
  float*          zb   = (float*)carve((size_t)(EE + NN) * 8 * 4);
  float*          deg  = (float*)carve((size_t)NN * 4);
  float*          sea  = (float*)carve((size_t)NN * 4);
  float*          la   = (float*)carve((size_t)NN * 4);
  float*          sv   = (float*)carve(64);
  float*          gsum = (float*)carve((size_t)Gg * 256 * 4);
  float*          gcnt = (float*)carve((size_t)Gg * 4);

  auto fill = [&](void* p, uint32_t v, int n) {
    k_fill<<<(n + 255) / 256, 256, 0, stream>>>((uint32_t*)p, v, n);
  };

  for (int l = 0; l < 3; ++l) {
    const bool loops = (l > 0);
    const int  Etot  = EE + (loops ? NN : 0);
    const float *W = Lp(l,0), *asr = Lp(l,1), *ads = Lp(l,2),
                *We = Lp(l,3), *aeg = Lp(l,4), *bb = Lp(l,5);

    if (loops) {  // self-loop attr = mean of incoming edge_attr per node
      fill(deg, 0u, NN);
      fill(sea, 0u, NN);
      k_degsum<<<(EE + 255) / 256, 256, 0, stream>>>(dst, ea, deg, sea, EE);
      k_loopattr<<<(NN + 255) / 256, 256, 0, stream>>>(deg, sea, la, NN);
    }

    // h = x @ W
    if (l == 0) {
      k_gemm_in4<<<NN, 256, 0, stream>>>(d_x, W, hbuf);
    } else {
      k_wt<<<256, 256, 0, stream>>>(W, wt);
      dim3 grid((NN + 63) / 64, 4);
      k_gemm_wmma<<<grid, 128, 0, stream>>>(abf, wt, hbuf, NN);
    }

    k_att<<<(NN * 8 + 255) / 256, 256, 0, stream>>>(hbuf, asr, ads, as_, ad_, NN);
    k_svec<<<1, 8, 0, stream>>>(We, aeg, sv);

    fill(menc, 0x807FFFFFu, NN * 8);      // fenc(-inf)
    fill(den,  0u, NN * 8);
    fill(xbuf, 0u, NN * 256);             // x buffer is dead after GEMM -> reuse as out

    k_logits <<<(Etot + 255) / 256, 256, 0, stream>>>(src, dst, ea, la, as_, ad_, sv,
                                                      zb, menc, EE, Etot - EE);
    k_expz   <<<(Etot + 255) / 256, 256, 0, stream>>>(dst, zb, menc, den, EE, Etot - EE);
    k_scatter<<<Etot, 256, 0, stream>>>(src, dst, zb, den, hbuf, xbuf, EE);
    k_biasrelu<<<(NN * 256 + 255) / 256, 256, 0, stream>>>(xbuf, bb, abf, NN * 256);
  }

  fill(gsum, 0u, Gg * 256);
  fill(gcnt, 0u, Gg);
  k_pool<<<NN, 256, 0, stream>>>(xbuf, bat, gsum, gcnt);
  k_head<<<Gg, 128, 0, stream>>>(gsum, gcnt, fc1W, fc1b, fc2W, fc2b, (float*)d_out);
}
